// GCN_57655640981899
// MI455X (gfx1250) — matrix operand order repaired and checked by reading, not compiled, not run on hardware
//
#include <hip/hip_runtime.h>
#include <hip/hip_bf16.h>

// ---------------------------------------------------------------------------
// EGNN forward for MI455X (gfx1250, wave32).
// Dominant cost: edge MLPs over B*N*N = 430,592 edges. We decompose the
// concat-GEMMs so the only edge-granularity GEMM is t @ eW2 (K=128), done
// with v_wmma_f32_16x16x32_f16 from LDS-staged f16 tiles, fused with the
// silu / edge-mask / sum_j aggregation so no [B,N,N,128] tensor hits HBM.
// Node-level GEMMs use a 16x64-per-wave WMMA tile (A-fragment reused 4x).
// ---------------------------------------------------------------------------

typedef __attribute__((ext_vector_type(16))) _Float16 v16h;
typedef __attribute__((ext_vector_type(8)))  _Float16 v8h;
typedef __attribute__((ext_vector_type(8)))  float    v8f;

#define NB 512          // batch (molecules)
#define NA 29           // atoms per molecule
#define BN (NB*NA)      // 14848, multiple of 16
#define HD 128          // hidden dim

union AFrag { v16h v; v8h h[2]; };

__device__ __forceinline__ float silu_f(float x) {
    return x * (1.0f / (1.0f + __expf(-x)));
}

// ---------------- prep kernels ----------------

__global__ void k_center(const float* __restrict__ pos, float* __restrict__ posc) {
    int b = blockIdx.x * blockDim.x + threadIdx.x;
    if (b >= NB) return;
    const float* p = pos + (size_t)b * NA * 3;
    float mx = 0.f, my = 0.f, mz = 0.f;
    for (int j = 0; j < NA; ++j) { mx += p[3*j]; my += p[3*j+1]; mz += p[3*j+2]; }
    mx /= NA; my /= NA; mz /= NA;
    float* o = posc + (size_t)b * NA * 3;
    for (int j = 0; j < NA; ++j) {
        o[3*j]   = p[3*j]   - mx;
        o[3*j+1] = p[3*j+1] - my;
        o[3*j+2] = p[3*j+2] - mz;
    }
}

__global__ void k_dist(const float* __restrict__ posc, const float* __restrict__ nm,
                       float* __restrict__ dist, float* __restrict__ em) {
    int bi = blockIdx.x;          // b*NA + i
    int j  = threadIdx.x;
    if (j >= NA) return;
    int b = bi / NA, i = bi % NA;
    const float* pi = posc + ((size_t)b * NA + i) * 3;
    const float* pj = posc + ((size_t)b * NA + j) * 3;
    float dx = pi[0]-pj[0], dy = pi[1]-pj[1], dz = pi[2]-pj[2];
    float d2 = dx*dx + dy*dy + dz*dz;
    float d  = (d2 > 0.f) ? sqrtf(d2) : 0.f;
    float e0 = nm[(size_t)b*NA + i] * nm[(size_t)b*NA + j];
    float dm = d * e0;                                 // dist = dist * edge_mask
    float e1 = e0 * ((dm <= 5.0f) ? 1.f : 0.f);        // hard cutoff
    dist[(size_t)bi*NA + j] = dm;
    em  [(size_t)bi*NA + j] = e1;
}

__global__ void k_embed(const int* __restrict__ z, const float* __restrict__ emb,
                        _Float16* __restrict__ h0) {
    int bi = blockIdx.x; int k = threadIdx.x;          // 32 threads (K padded 20->32)
    int zi = z[bi];
    float v = (k < 20) ? emb[(size_t)zi*20 + k] : 0.f;
    h0[(size_t)bi*32 + k] = (_Float16)v;
}

// f32 [K x Nc] (optionally row-offset slice) -> transposed, zero-padded f16 [Nc x Kpad]
__global__ void k_cvt_t(const float* __restrict__ src, _Float16* __restrict__ dst,
                        int K, int Nc, int Kpad, int roff) {
    int t = blockIdx.x * blockDim.x + threadIdx.x;
    if (t >= Nc * Kpad) return;
    int n = t / Kpad, k = t % Kpad;
    float v = (k < K) ? src[(size_t)(roff + k) * Nc + n] : 0.f;
    dst[(size_t)n * Kpad + k] = (_Float16)v;
}

// ---------------- generic WMMA GEMM (16x64 per wave) ----------------
// out[M,Nc] = A[M,K](f16) @ Bt[Nc,K]^T(f16)  (+Cin)(+bias)(silu?)(*rowmask)
// One wave per 16(M) x 64(N) tile: A fragment loaded once per K-chunk and
// reused across 4 B fragments (4 accumulators) for 4x better A reuse.
__global__ void __launch_bounds__(32)
k_gemm(const _Float16* __restrict__ A, const _Float16* __restrict__ Bt,
       const float* __restrict__ bias, const float* __restrict__ Cin,
       const float* __restrict__ rowmask,
       float* __restrict__ outF, _Float16* __restrict__ outH,
       int Nc, int K, int do_silu) {
    int m0 = blockIdx.x * 16, n0 = blockIdx.y * 64;
    int lane = threadIdx.x;
    int r16 = lane & 15, hi = lane >> 4;
    v8f acc0 = {}, acc1 = {}, acc2 = {}, acc3 = {};
    for (int k0 = 0; k0 < K; k0 += 32) {
        // A 16x32 f16 frag: lanes 0-15 row m, K {0..15}; lanes 16-31 K {16..31}
        AFrag a;
        const _Float16* ar = A + (size_t)(m0 + r16) * K + k0 + hi * 16;
        a.h[0] = *(const v8h*)(ar);
        a.h[1] = *(const v8h*)(ar + 8);
        if (k0 + 32 < K) __builtin_prefetch(ar + 32, 0, 3);   // global_prefetch_b8
        // B 32x16 frags: lane = column, 16 contiguous K values per lane-half
        const _Float16* br = Bt + (size_t)(n0 + r16) * K + k0 + hi * 16;
        v16h b0 = *(const v16h*)(br);
        v16h b1 = *(const v16h*)(br + 16 * K);
        v16h b2 = *(const v16h*)(br + 32 * K);
        v16h b3 = *(const v16h*)(br + 48 * K);
        acc0 = __builtin_amdgcn_wmma_f32_16x16x32_f16(false, a.v, false, b0, (short)0, acc0, false, false);
        acc1 = __builtin_amdgcn_wmma_f32_16x16x32_f16(false, a.v, false, b1, (short)0, acc1, false, false);
        acc2 = __builtin_amdgcn_wmma_f32_16x16x32_f16(false, a.v, false, b2, (short)0, acc2, false, false);
        acc3 = __builtin_amdgcn_wmma_f32_16x16x32_f16(false, a.v, false, b3, (short)0, acc3, false, false);
    }
    v8f accs[4] = {acc0, acc1, acc2, acc3};
    for (int t = 0; t < 4; ++t) {
        int col = n0 + t * 16 + r16;
        float bv = bias ? bias[col] : 0.f;
        for (int r = 0; r < 8; ++r) {
            int m = m0 + r + 8 * hi;   // C layout: VGPR r = row r (lanes 0-15) / 8+r (16-31)
            float v = accs[t][r];
            if (Cin) v += Cin[(size_t)m * Nc + col];
            v += bv;
            if (do_silu) v = silu_f(v);
            if (rowmask) v *= rowmask[m];
            if (outF) outF[(size_t)m * Nc + col] = v;
            if (outH) outH[(size_t)m * Nc + col] = (_Float16)v;
        }
    }
}

// ---------------- fused edge MLP + aggregation ----------------
// One block per (b,i). t[j,k] = silu(P[b,i,k] + Q[b,j,k] + d(i,j)*w1c[k] + eb1[k])
// agg[b,i,:] = sum_j silu(t @ eW2 + eb2)[j,:] * em[b,i,j]
__global__ void __launch_bounds__(256)
k_edge(const float* __restrict__ P, const float* __restrict__ Q,
       const float* __restrict__ dist, const float* __restrict__ em,
       const float* __restrict__ w1c, const float* __restrict__ eb1,
       const _Float16* __restrict__ W2t, const float* __restrict__ eb2,
       float* __restrict__ aggF, _Float16* __restrict__ aggH) {
    __shared__ float Pi[HD];
    __shared__ float wc[HD];
    __shared__ float emL[32];
    __shared__ __align__(32) _Float16 W2s[HD * HD];   // 32 KB, transposed weights
    __shared__ __align__(32) _Float16 T[32 * HD];     // 8 KB, current 16-row j-tile

    int tid = threadIdx.x;
    int bi  = blockIdx.x;
    int b   = bi / NA;

    if (tid < HD) { Pi[tid] = P[(size_t)bi * HD + tid] + eb1[tid]; wc[tid] = w1c[tid]; }
    for (int off = tid * 8; off < HD * HD; off += 256 * 8)
        *(v8h*)(W2s + off) = *(const v8h*)(W2t + off);
    // warm L2/L0 for the second j-chunk's Q rows while we work on the first
    {
        int jp = 16 + (tid >> 4);
        if (jp < NA) __builtin_prefetch(Q + ((size_t)b * NA + jp) * HD + (tid & 15) * 8, 0, 3);
    }
    __syncthreads();

    int lane = tid & 31, w = tid >> 5;   // 8 waves, wave w owns output cols [16w,16w+16)
    int r16 = lane & 15, hi = lane >> 4;
    int n0  = w * 16;
    float partial = 0.f;

    for (int c = 0; c < 2; ++c) {        // j in [0,16) then [16,32) (29 padded to 32)
        int j0 = c * 16;
        {   // cooperative fill of T: 16 threads per row, 8 k each
            int jj = tid >> 4;
            int kb = (tid & 15) * 8;
            int j  = j0 + jj;
            if (j < NA) {
                const float* qrow = Q + ((size_t)b * NA + j) * HD;
                float d = dist[(size_t)bi * NA + j];
                for (int e = 0; e < 8; ++e) {
                    int k = kb + e;
                    T[jj * HD + k] = (_Float16)silu_f(Pi[k] + qrow[k] + d * wc[k]);
                }
            } else {
                for (int e = 0; e < 8; ++e) T[jj * HD + kb + e] = (_Float16)0.f;
            }
            if (tid < 16) emL[tid] = (j0 + tid < NA) ? em[(size_t)bi * NA + j0 + tid] : 0.f;
        }
        __syncthreads();

        v8f acc = {};
        for (int k0 = 0; k0 < HD; k0 += 32) {
            AFrag a;                                  // A frag from LDS (ds_load_b128 x2)
            const _Float16* ar = &T[r16 * HD + k0 + hi * 16];
            a.h[0] = *(const v8h*)(ar);
            a.h[1] = *(const v8h*)(ar + 8);
            v16h bb = *(const v16h*)(&W2s[(n0 + r16) * HD + k0 + hi * 16]);
            acc = __builtin_amdgcn_wmma_f32_16x16x32_f16(false, a.v, false, bb,
                                                         (short)0, acc, false, false);
        }
        int col = n0 + r16;
        float bv = eb2[col];
        for (int r = 0; r < 8; ++r) {
            int mrow = r + 8 * hi;                    // j index within chunk
            partial += silu_f(acc[r] + bv) * emL[mrow];
        }
        __syncthreads();
    }
    // lanes l and l+16 hold complementary row-halves of the same column
    float tot = partial + __shfl_xor(partial, 16, 32);
    if (lane < 16) {
        int col = n0 + lane;
        aggF[(size_t)bi * HD + col] = tot;
        aggH[(size_t)bi * HD + col] = (_Float16)tot;
    }
}

// ---------------- readout tail ----------------

__global__ void k_zero(float* __restrict__ p, int n) {
    int t = blockIdx.x * blockDim.x + threadIdx.x;
    if (t < n) p[t] = 0.f;
}

__global__ void k_dot(const _Float16* __restrict__ r1, const float* __restrict__ oW2,
                      const float* __restrict__ ob2, const float* __restrict__ nm,
                      float* __restrict__ outm) {
    int a = blockIdx.x * blockDim.x + threadIdx.x;
    if (a >= BN) return;
    const _Float16* row = r1 + (size_t)a * 64;
    float s = ob2[0];
    for (int e = 0; e < 64; ++e) s += (float)row[e] * oW2[e];
    s *= nm[a];
    atomicAdd(&outm[a / NA], s);
}

__global__ void k_loss(const float* __restrict__ outm, const float* __restrict__ u0,
                       float* __restrict__ out2) {
    __shared__ float se[256], sa[256];
    int t = threadIdx.x;
    float e2 = 0.f, ea = 0.f;
    for (int b = t; b < NB; b += 256) {
        float d = outm[b] - u0[b];
        e2 += d * d;
        ea += fabsf(d);
    }
    se[t] = e2; sa[t] = ea;
    __syncthreads();
    for (int s = 128; s > 0; s >>= 1) {
        if (t < s) { se[t] += se[t + s]; sa[t] += sa[t + s]; }
        __syncthreads();
    }
    if (t == 0) {
        out2[0] = sqrtf(se[0] / NB);
        out2[1] = sa[0] / NB;
    }
}

// ---------------- host orchestration ----------------

extern "C" void kernel_launch(void* const* d_in, const int* in_sizes, int n_in,
                              void* d_out, int out_size, void* d_ws, size_t ws_size,
                              hipStream_t stream) {
    (void)in_sizes; (void)out_size; (void)ws_size;
    if (n_in < 49) return;   // expect 4 inputs + 45 flattened params (insertion order)

    const int*   zc  = (const int*)  d_in[0];   // atomic_numbers [B,N]
    const float* pos = (const float*)d_in[1];   // positions [B,N,3]
    const float* nm  = (const float*)d_in[2];   // atom_mask [B,N]
    const float* u0  = (const float*)d_in[3];   // u0 [B,1]
    const float* emb = (const float*)d_in[4];   // emb [10,20]
    const float* oW1 = (const float*)d_in[45];  // [128,64]
    const float* ob1 = (const float*)d_in[46];
    const float* oW2 = (const float*)d_in[47];  // [64,1]
    const float* ob2 = (const float*)d_in[48];

    char* wp = (char*)d_ws;
    auto alloc = [&](size_t bytes) -> char* {
        char* r = wp; wp += (bytes + 255) & ~(size_t)255; return r;
    };

    float*    posc = (float*)   alloc((size_t)BN * 3  * sizeof(float));
    float*    dist = (float*)   alloc((size_t)BN * NA * sizeof(float));
    float*    em   = (float*)   alloc((size_t)BN * NA * sizeof(float));
    _Float16* h0   = (_Float16*)alloc((size_t)BN * 32 * sizeof(_Float16));
    _Float16* hA   = (_Float16*)alloc((size_t)BN * HD * sizeof(_Float16));
    _Float16* hB   = (_Float16*)alloc((size_t)BN * HD * sizeof(_Float16));
    float*    Pf   = (float*)   alloc((size_t)BN * HD * sizeof(float));
    float*    Qf   = (float*)   alloc((size_t)BN * HD * sizeof(float));
    float*    aggF = (float*)   alloc((size_t)BN * HD * sizeof(float));
    _Float16* aggH = (_Float16*)alloc((size_t)BN * HD * sizeof(_Float16));
    float*    tmpF = (float*)   alloc((size_t)BN * HD * sizeof(float));
    _Float16* n1h  = (_Float16*)alloc((size_t)BN * HD * sizeof(_Float16));
    _Float16* r1h  = (_Float16*)alloc((size_t)BN * 64 * sizeof(_Float16));
    float*    outm = (float*)   alloc((size_t)NB * sizeof(float));
    _Float16* w1a  = (_Float16*)alloc((size_t)HD * HD * sizeof(_Float16));
    _Float16* w1b  = (_Float16*)alloc((size_t)HD * HD * sizeof(_Float16));
    _Float16* w2t  = (_Float16*)alloc((size_t)HD * HD * sizeof(_Float16));
    _Float16* wn1a = (_Float16*)alloc((size_t)HD * HD * sizeof(_Float16));
    _Float16* wn1b = (_Float16*)alloc((size_t)HD * HD * sizeof(_Float16));
    _Float16* wn2t = (_Float16*)alloc((size_t)HD * HD * sizeof(_Float16));
    _Float16* ow1t = (_Float16*)alloc((size_t)64 * HD * sizeof(_Float16));

    auto cvt = [&](const float* src, _Float16* dst, int K, int Nc, int Kpad, int roff) {
        int tot = Nc * Kpad;
        k_cvt_t<<<(tot + 255) / 256, 256, 0, stream>>>(src, dst, K, Nc, Kpad, roff);
    };

    k_center<<<(NB + 255) / 256, 256, 0, stream>>>(pos, posc);
    k_dist<<<BN, 32, 0, stream>>>(posc, nm, dist, em);
    k_embed<<<BN, 32, 0, stream>>>(zc, emb, h0);

    const _Float16* hin = h0;
    for (int l = 0; l < 5; ++l) {
        int Fsrc = (l == 0) ? 20 : 128;   // real input feature dim
        int Kf   = (l == 0) ? 32 : 128;   // K padded to multiple of 32
        const float* eW1 = (const float*)d_in[5 + 8 * l];
        const float* eb1 = (const float*)d_in[6 + 8 * l];
        const float* eW2 = (const float*)d_in[7 + 8 * l];
        const float* eb2 = (const float*)d_in[8 + 8 * l];
        const float* nW1 = (const float*)d_in[9 + 8 * l];
        const float* nb1 = (const float*)d_in[10 + 8 * l];
        const float* nW2 = (const float*)d_in[11 + 8 * l];
        const float* nb2 = (const float*)d_in[12 + 8 * l];
        const float* w1c = eW1 + (size_t)(2 * Fsrc) * HD;   // distance row of eW1

        cvt(eW1, w1a,  Fsrc, HD, Kf,  0);      // h_i part of eW1
        cvt(eW1, w1b,  Fsrc, HD, Kf,  Fsrc);   // h_j part of eW1
        cvt(eW2, w2t,  HD,   HD, HD,  0);
        cvt(nW1, wn1a, Fsrc, HD, Kf,  0);      // h part of nW1
        cvt(nW1, wn1b, HD,   HD, HD,  Fsrc);   // agg part of nW1
        cvt(nW2, wn2t, HD,   HD, HD,  0);

        dim3 g(BN / 16, HD / 64);   // 16x64 tile per wave
        // node-level pre-projections: P = h@W1a, Q = h@W1b  (f32 out)
        k_gemm<<<g, 32, 0, stream>>>(hin, w1a, nullptr, nullptr, nullptr, Pf, nullptr, HD, Kf, 0);
        k_gemm<<<g, 32, 0, stream>>>(hin, w1b, nullptr, nullptr, nullptr, Qf, nullptr, HD, Kf, 0);
        // fused edge MLP + masked sum_j  ->  agg
        k_edge<<<BN, 256, 0, stream>>>(Pf, Qf, dist, em, w1c, eb1, w2t, eb2, aggF, aggH);
        // node update: h = silu(h@Wn1a + agg@Wn1b + nb1) @ nW2 + nb2, masked
        k_gemm<<<g, 32, 0, stream>>>(hin, wn1a, nullptr, nullptr, nullptr, tmpF, nullptr, HD, Kf, 0);
        k_gemm<<<g, 32, 0, stream>>>(aggH, wn1b, nb1, tmpF, nullptr, nullptr, n1h, HD, HD, 1);
        _Float16* hout = (l & 1) ? hB : hA;
        k_gemm<<<g, 32, 0, stream>>>(n1h, wn2t, nb2, nullptr, nm, nullptr, hout, HD, HD, 0);
        hin = hout;
    }

    // readout: r1 = silu(h@oW1 + ob1) [BN,64]; out = r1@oW2 + ob2, masked sum-pool
    cvt(oW1, ow1t, HD, 64, HD, 0);
    dim3 gr(BN / 16, 1);            // Nc=64 -> one 16x64 tile per wave
    k_gemm<<<gr, 32, 0, stream>>>(hin, ow1t, ob1, nullptr, nullptr, nullptr, r1h, 64, HD, 1);
    k_zero<<<(NB + 255) / 256, 256, 0, stream>>>(outm, NB);
    k_dot<<<(BN + 255) / 256, 256, 0, stream>>>(r1h, oW2, ob2, nm, outm);
    k_loss<<<1, 256, 0, stream>>>(outm, u0, (float*)d_out);
}
